// MultiHeadAttention_28793460753189
// MI455X (gfx1250) — matrix operand order repaired and checked by reading, not compiled, not run on hardware
//
#include <hip/hip_runtime.h>
#include <hip/hip_bf16.h>

// ---------------------------------------------------------------------------
// MHA forward for MI455X (gfx1250, wave32, WMMA).
//   fp32->bf16 convert -> 3x projection GEMM (bf16 WMMA, f32 acc; Q prescaled
//   by 0.125*log2e) -> flash-attention (exp2-domain online softmax, 64-key
//   tiles) -> output projection GEMM (f32 out).
// ---------------------------------------------------------------------------

#define Bb     2
#define Lseq   2048
#define Dmodel 1024
#define Hh     16
#define HD     64
#define Mrows  (Bb * Lseq)   // 4096

typedef __attribute__((ext_vector_type(16))) __bf16        bf16x16;
typedef __attribute__((ext_vector_type(8)))  float         f32x8;
typedef __attribute__((ext_vector_type(8)))  unsigned int  u32x8;

__device__ __forceinline__ unsigned short f32_to_bf16(float f) {
    unsigned int u = __float_as_uint(f);
    u += 0x7fffu + ((u >> 16) & 1u);          // round to nearest even
    return (unsigned short)(u >> 16);
}

__device__ __forceinline__ f32x8 wmma_bf16(bf16x16 a, bf16x16 b, f32x8 c) {
    return __builtin_amdgcn_wmma_f32_16x16x32_bf16(
        false, a, false, b, (short)0, c, false, false);
}

// A-fragment K index for 16x32 bf16 A layout (row = lane%16, half = lane/16)
__device__ __forceinline__ int a_kk(int v, int half) {
    return (v < 4) ? (half * 8 + v * 2) : (16 + half * 8 + (v - 4) * 2);
}

// ---------------------------------------------------------------------------
// fp32 -> bf16 streaming convert (4 elements / thread)
// ---------------------------------------------------------------------------
__global__ __launch_bounds__(256)
void cvt_f32_bf16_kernel(const float* __restrict__ src,
                         unsigned short* __restrict__ dst, int n) {
    int i = (blockIdx.x * 256 + threadIdx.x) * 4;
    if (i < n) {
        float4 f = *(const float4*)(src + i);
        uint2 o;
        o.x = (unsigned int)f32_to_bf16(f.x) | ((unsigned int)f32_to_bf16(f.y) << 16);
        o.y = (unsigned int)f32_to_bf16(f.z) | ((unsigned int)f32_to_bf16(f.w) << 16);
        *(uint2*)(dst + i) = o;
    }
}

// ---------------------------------------------------------------------------
// GEMM: out[M,N] = scale * (A[M,K](bf16) @ W[N,K]^T + bias)
// Block 256 thr = 8 waves (2 M x 4 N); wave tile 64x64 (16 accumulators).
// Block tile 128(M) x 256(N); A and W tiles staged in LDS per 32-k step.
// ---------------------------------------------------------------------------
template <bool F32OUT>
__global__ __launch_bounds__(256)
void gemm_bias_kernel(const unsigned short* __restrict__ A,
                      const unsigned short* __restrict__ W,
                      const float* __restrict__ bias,
                      unsigned short* __restrict__ outB,
                      float* __restrict__ outF,
                      float scale) {
    constexpr int Ktot = Dmodel, Ntot = Dmodel;
    __shared__ __attribute__((aligned(16))) unsigned short At[128 * 32]; // [m][k]
    __shared__ __attribute__((aligned(16))) unsigned short Wt[256 * 32]; // [n][k]

    const int lane = threadIdx.x & 31;
    const int wave = threadIdx.x >> 5;
    const int half = lane >> 4;
    const int l16  = lane & 15;
    const int wm   = wave >> 2;            // 0..1
    const int wn   = wave & 3;             // 0..3
    const int m0   = blockIdx.y * 128;
    const int n0   = blockIdx.x * 256;

    f32x8 acc[4][4] = {};                  // [mi][j]

    for (int k0 = 0; k0 < Ktot; k0 += 32) {
        // stage A tile (128x32): 2 b128 per thread
#pragma unroll
        for (int i = 0; i < 2; ++i) {
            int idx = threadIdx.x + 256 * i;
            int ar = idx >> 2, ak = (idx & 3) * 8;
            *(uint4*)(&At[ar * 32 + ak]) =
                *(const uint4*)(A + (size_t)(m0 + ar) * Ktot + k0 + ak);
        }
        // stage W tile (256x32): 4 b128 per thread
#pragma unroll
        for (int i = 0; i < 4; ++i) {
            int idx = threadIdx.x + 256 * i;
            int wr = idx >> 2, wk = (idx & 3) * 8;
            *(uint4*)(&Wt[wr * 32 + wk]) =
                *(const uint4*)(W + (size_t)(n0 + wr) * Ktot + k0 + wk);
        }
        __syncthreads();

        // A fragments (4 x 16 rows)
        bf16x16 af[4];
#pragma unroll
        for (int mi = 0; mi < 4; ++mi) {
            int row = wm * 64 + mi * 16 + l16;
            u32x8 au;
#pragma unroll
            for (int v = 0; v < 8; ++v)
                au[v] = *(const unsigned int*)(&At[row * 32 + a_kk(v, half)]);
            af[mi] = __builtin_bit_cast(bf16x16, au);
        }
        // B fragments (4 x 16 cols), each reused across the 4 M sub-tiles
#pragma unroll
        for (int j = 0; j < 4; ++j) {
            int ncol = wn * 64 + j * 16 + l16;
            u32x8 bu;
#pragma unroll
            for (int v = 0; v < 8; ++v)
                bu[v] = *(const unsigned int*)(&Wt[ncol * 32 + a_kk(v, half)]);
            bf16x16 bf = __builtin_bit_cast(bf16x16, bu);
#pragma unroll
            for (int mi = 0; mi < 4; ++mi)
                acc[mi][j] = wmma_bf16(af[mi], bf, acc[mi][j]);
        }
        __syncthreads();
    }

    // epilogue: scale*(acc + bias), store (C: row = v + 8*half, col = lane%16)
#pragma unroll
    for (int j = 0; j < 4; ++j) {
        int n = n0 + wn * 64 + j * 16 + l16;
        float bv = bias[n];
#pragma unroll
        for (int mi = 0; mi < 4; ++mi) {
#pragma unroll
            for (int v = 0; v < 8; ++v) {
                int m = m0 + wm * 64 + mi * 16 + v + 8 * half;
                float val = (acc[mi][j][v] + bv) * scale;
                if (F32OUT) outF[(size_t)m * Ntot + n] = val;
                else        outB[(size_t)m * Ntot + n] = f32_to_bf16(val);
            }
        }
    }
}

// ---------------------------------------------------------------------------
// Flash attention: block = (b, h, 128 query rows); 8 waves x 16 rows.
// 64-key tiles through LDS; exp2-domain online softmax (Q prescaled).
// ---------------------------------------------------------------------------
__global__ __launch_bounds__(256)
void attention_kernel(const unsigned short* __restrict__ Qb,
                      const unsigned short* __restrict__ Kb,
                      const unsigned short* __restrict__ Vb,
                      unsigned short* __restrict__ Ob) {
    __shared__ __attribute__((aligned(16))) unsigned short Kt[64 * 64];     // [key][dim]
    __shared__ __attribute__((aligned(16))) unsigned short Vt[64 * 64];     // [dim][key]
    __shared__ __attribute__((aligned(16))) unsigned short Ps[8 * 16 * 64]; // per-wave P

    const int lane = threadIdx.x & 31;
    const int wave = threadIdx.x >> 5;
    const int half = lane >> 4;
    const int l16  = lane & 15;
    const int bh   = blockIdx.y;          // 0..31
    const int b    = bh >> 4;
    const int h    = bh & 15;
    const int q0   = blockIdx.x * 128 + wave * 16;
    const size_t baseQ = ((size_t)(b * Lseq + q0 + l16)) * Dmodel + h * HD;

    // Q fragments (prescaled by 0.125*log2e at projection time)
    const unsigned int* Q32 = (const unsigned int*)Qb;
    bf16x16 qf[2];
#pragma unroll
    for (int kk2 = 0; kk2 < 2; ++kk2) {
        u32x8 qu;
#pragma unroll
        for (int v = 0; v < 8; ++v)
            qu[v] = Q32[(baseQ + kk2 * 32 + a_kk(v, half)) >> 1];
        qf[kk2] = __builtin_bit_cast(bf16x16, qu);
    }

    f32x8 O[4] = {};
    float Mst[8], Lst[8];
#pragma unroll
    for (int v = 0; v < 8; ++v) { Mst[v] = -1e30f; Lst[v] = 0.f; }

    unsigned short* Pw = &Ps[wave * (16 * 64)];

    for (int kv0 = 0; kv0 < Lseq; kv0 += 64) {
        // --- cooperative stage: K row-major [key][dim], V transposed [dim][key]
#pragma unroll
        for (int i = 0; i < 2; ++i) {
            int key = i * 32 + (threadIdx.x >> 3);
            int d0  = (threadIdx.x & 7) * 8;
            size_t g = ((size_t)(b * Lseq + kv0 + key)) * Dmodel + h * HD + d0;
            *(uint4*)(&Kt[key * 64 + d0]) = *(const uint4*)(Kb + g);
            uint4 vv = *(const uint4*)(Vb + g);
            unsigned int w[4] = {vv.x, vv.y, vv.z, vv.w};
#pragma unroll
            for (int u = 0; u < 4; ++u) {
                Vt[(d0 + 2 * u)     * 64 + key] = (unsigned short)(w[u] & 0xffffu);
                Vt[(d0 + 2 * u + 1) * 64 + key] = (unsigned short)(w[u] >> 16);
            }
        }
        __syncthreads();

        // --- S = Q K^T : four 16x16 tiles (already in log2 domain) ---
        f32x8 s[4];
#pragma unroll
        for (int tt = 0; tt < 4; ++tt) {
            f32x8 c = {};
            int key = tt * 16 + l16;
#pragma unroll
            for (int kk2 = 0; kk2 < 2; ++kk2) {
                u32x8 bu;
#pragma unroll
                for (int v = 0; v < 8; ++v)
                    bu[v] = *(const unsigned int*)(&Kt[key * 64 + kk2 * 32 + a_kk(v, half)]);
                c = wmma_bf16(qf[kk2], __builtin_bit_cast(bf16x16, bu), c);
            }
            s[tt] = c;
        }

        // --- online softmax in exp2 domain (row r = v + 8*half) ---
#pragma unroll
        for (int v = 0; v < 8; ++v) {
            float mt = fmaxf(fmaxf(s[0][v], s[1][v]), fmaxf(s[2][v], s[3][v]));
            mt = fmaxf(mt, __shfl_xor(mt, 1, 16));
            mt = fmaxf(mt, __shfl_xor(mt, 2, 16));
            mt = fmaxf(mt, __shfl_xor(mt, 4, 16));
            mt = fmaxf(mt, __shfl_xor(mt, 8, 16));
            float Mn = fmaxf(Mst[v], mt);
            float al = __builtin_amdgcn_exp2f(Mst[v] - Mn);
            float rs = 0.f;
#pragma unroll
            for (int tt = 0; tt < 4; ++tt) {
                float e = __builtin_amdgcn_exp2f(s[tt][v] - Mn);
                s[tt][v] = e;
                rs += e;
            }
            rs += __shfl_xor(rs, 1, 16);
            rs += __shfl_xor(rs, 2, 16);
            rs += __shfl_xor(rs, 4, 16);
            rs += __shfl_xor(rs, 8, 16);
            Lst[v] = Lst[v] * al + rs;
            Mst[v] = Mn;
#pragma unroll
            for (int j = 0; j < 4; ++j) O[j][v] *= al;
        }

        // --- transpose P (C layout -> A layout) through per-wave LDS ---
#pragma unroll
        for (int tt = 0; tt < 4; ++tt)
#pragma unroll
            for (int v = 0; v < 8; ++v)
                Pw[(v + 8 * half) * 64 + tt * 16 + l16] = f32_to_bf16(s[tt][v]);
        __builtin_amdgcn_wave_barrier();   // LDS in-order per wave
        bf16x16 pf[2];
#pragma unroll
        for (int pk = 0; pk < 2; ++pk) {
            u32x8 pu;
#pragma unroll
            for (int v = 0; v < 8; ++v)
                pu[v] = *(const unsigned int*)(&Pw[l16 * 64 + pk * 32 + a_kk(v, half)]);
            pf[pk] = __builtin_bit_cast(bf16x16, pu);
        }

        // --- O += P (16x64) x V (64x64) ---
#pragma unroll
        for (int j = 0; j < 4; ++j) {
            int d = j * 16 + l16;
#pragma unroll
            for (int pk = 0; pk < 2; ++pk) {
                u32x8 bu;
#pragma unroll
                for (int v = 0; v < 8; ++v)
                    bu[v] = *(const unsigned int*)(&Vt[d * 64 + pk * 32 + a_kk(v, half)]);
                O[j] = wmma_bf16(pf[pk], __builtin_bit_cast(bf16x16, bu), O[j]);
            }
        }
        __syncthreads();
    }

    // --- normalize (one reciprocal per row) and store bf16 ---
    float rl[8];
#pragma unroll
    for (int v = 0; v < 8; ++v) rl[v] = 1.0f / Lst[v];
#pragma unroll
    for (int j = 0; j < 4; ++j)
#pragma unroll
        for (int v = 0; v < 8; ++v) {
            int row = q0 + v + 8 * half;
            Ob[((size_t)(b * Lseq + row)) * Dmodel + h * HD + j * 16 + l16] =
                f32_to_bf16(O[j][v] * rl[v]);
        }
}

// ---------------------------------------------------------------------------
// Launch
// ---------------------------------------------------------------------------
extern "C" void kernel_launch(void* const* d_in, const int* in_sizes, int n_in,
                              void* d_out, int out_size, void* d_ws, size_t ws_size,
                              hipStream_t stream) {
    const float* query = (const float*)d_in[0];
    const float* key_  = (const float*)d_in[1];
    const float* value = (const float*)d_in[2];
    const float* Wq = (const float*)d_in[3];
    const float* bq = (const float*)d_in[4];
    const float* Wk = (const float*)d_in[5];
    const float* bk = (const float*)d_in[6];
    const float* Wv = (const float*)d_in[7];
    const float* bv = (const float*)d_in[8];
    const float* Wo = (const float*)d_in[9];
    const float* bo = (const float*)d_in[10];
    float* out = (float*)d_out;

    const size_t MD = (size_t)Mrows * Dmodel;
    const size_t DD = (size_t)Dmodel * Dmodel;
    unsigned short* ws  = (unsigned short*)d_ws;
    unsigned short* Xq  = ws;
    unsigned short* Xk  = Xq  + MD;
    unsigned short* Xv  = Xk  + MD;
    unsigned short* Wqb = Xv  + MD;
    unsigned short* Wkb = Wqb + DD;
    unsigned short* Wvb = Wkb + DD;
    unsigned short* Wob = Wvb + DD;
    unsigned short* Qb  = Wob + DD;
    unsigned short* Kb  = Qb  + MD;
    unsigned short* Vb  = Kb  + MD;
    unsigned short* Ao  = Vb  + MD;

    auto cvt = [&](const float* s, unsigned short* d, size_t n) {
        cvt_f32_bf16_kernel<<<dim3((unsigned)((n / 4 + 255) / 256)), 256, 0, stream>>>(
            s, d, (int)n);
    };
    cvt(query, Xq, MD);
    cvt(key_,  Xk, MD);
    cvt(value, Xv, MD);
    cvt(Wq, Wqb, DD);
    cvt(Wk, Wkb, DD);
    cvt(Wv, Wvb, DD);
    cvt(Wo, Wob, DD);

    // scores/softmax run in exp2 domain: fold 1/sqrt(HD) * log2(e) into Q
    const float qscale = 0.125f * 1.44269504088896340736f;

    dim3 ggrid(Dmodel / 256, Mrows / 128);  // (4, 32)
    gemm_bias_kernel<false><<<ggrid, 256, 0, stream>>>(Xq, Wqb, bq, Qb, nullptr, qscale);
    gemm_bias_kernel<false><<<ggrid, 256, 0, stream>>>(Xk, Wkb, bk, Kb, nullptr, 1.0f);
    gemm_bias_kernel<false><<<ggrid, 256, 0, stream>>>(Xv, Wvb, bv, Vb, nullptr, 1.0f);

    attention_kernel<<<dim3(Lseq / 128, Bb * Hh), 256, 0, stream>>>(Qb, Kb, Vb, Ao);

    gemm_bias_kernel<true><<<ggrid, 256, 0, stream>>>(Ao, Wob, bo, nullptr, out, 1.0f);
}
